// LinearReader_335007450003
// MI455X (gfx1250) — matrix operand order repaired and checked by reading, not compiled, not run on hardware
//
#include <hip/hip_runtime.h>

// ---------------------------------------------------------------------------
// linear attention over external memory:
//   kv  = keys^T @ vals        [1024 x 1024]   (GEMM1: 137 GFLOP, 512 MB in)
//   out = x @ kv               [8192 x 1024]   (GEMM2: 17 GFLOP)
// bf16 WMMA / fp32 accumulate: fp32 WMMA (16x16x4) would sit ~8x above the
// 22us HBM floor for GEMM1; bf16 (16x16x32) is balanced (~55us compute).
// ---------------------------------------------------------------------------

typedef __attribute__((ext_vector_type(4)))  float          f32x4;
typedef __attribute__((ext_vector_type(8)))  float          f32x8;
typedef __attribute__((ext_vector_type(2)))  unsigned       u32x2;
typedef __attribute__((ext_vector_type(4)))  unsigned       u32x4;
typedef __attribute__((ext_vector_type(4)))  unsigned short u16x4;
typedef __attribute__((ext_vector_type(8)))  unsigned short u16x8;
typedef __attribute__((ext_vector_type(2)))  __bf16         bf16x2;
typedef __attribute__((ext_vector_type(16))) __bf16         bf16x16;

union Frag { bf16x16 bf; u16x8 h[2]; };

#define D_DIM       1024
#define M_DIM       65536
#define ROWS_X      8192            // B*S = 4*2048
#define LDS_STRIDE  64              // u16 per row (128 B), 8 groups of 16 B
#define BUF_OFS     (128 * LDS_STRIDE)   // u16 offset between ping-pong bufs

// packed fp32x2 -> bf16x2 (hardware RNE convert: v_cvt_pk_bf16_f32)
__device__ __forceinline__ unsigned pkbf(float a, float b) {
#if __has_builtin(__builtin_amdgcn_cvt_pk_bf16_f32)
  bf16x2 v = __builtin_amdgcn_cvt_pk_bf16_f32(a, b);
  return __builtin_bit_cast(unsigned, v);
#else
  unsigned r;
  asm("v_cvt_pk_bf16_f32 %0, %1, %2" : "=v"(r) : "v"(a), "v"(b));
  return r;
#endif
}

// bank-conflict swizzle: permute the 8 16-byte k-groups per row
__device__ __forceinline__ int swz(int row, int g) {
  return ((g ^ (row & 7)) + (row >> 3)) & 7;
}

// ===========================================================================
// Kernel 1: kv[d][e] = sum_m keys[m][d] * vals[m][e]   (output stored bf16)
// grid (8,8), block 256 (8 waves, 4x2). 128x128 tile, K staged 64 at a time.
// ===========================================================================
__global__ void __launch_bounds__(256)
kv_gemm_bf16(const float* __restrict__ keys,
             const float* __restrict__ vals,
             unsigned short* __restrict__ kvb)
{
  __shared__ alignas(16) unsigned short lA[2][BUF_OFS]; // keys^T tile
  __shared__ alignas(16) unsigned short lB[2][BUF_OFS]; // vals   tile

  const int t    = threadIdx.x;
  const int L    = t & 31;
  const int half = L >> 4;
  const int w    = t >> 5;
  const int wr   = w >> 1;            // 0..3 -> 32 d-rows each
  const int wc   = w & 1;             // 0..1 -> 64 e-cols each
  const int D0   = blockIdx.x * 128;
  const int E0   = blockIdx.y * 128;

  // staging: thread loads [8 k] x [4 cols], transposes in registers
  const int dq = t & 31;              // cols dq*4 .. dq*4+3
  const int kb = t >> 5;              // k = kb*8 .. kb*8+7

  // ---- loop-invariant LDS fragment pointers (buffer 0; buffer 1 = +BUF_OFS,
  //      which folds into the DS instruction's 16-bit immediate offset) ----
  const unsigned short *aP[2][2][2], *bP[2][4][2];
  #pragma unroll
  for (int kc = 0; kc < 2; ++kc) {
    #pragma unroll
    for (int i = 0; i < 2; ++i) {
      const int row = wr * 32 + i * 16 + (L & 15);
      aP[kc][i][0] = &lA[0][row * LDS_STRIDE + swz(row, kc * 4 + half) * 8];
      aP[kc][i][1] = &lA[0][row * LDS_STRIDE + swz(row, kc * 4 + 2 + half) * 8];
    }
    #pragma unroll
    for (int j = 0; j < 4; ++j) {
      const int row = wc * 64 + j * 16 + (L & 15);
      bP[kc][j][0] = &lB[0][row * LDS_STRIDE + swz(row, kc * 4 + half) * 8];
      bP[kc][j][1] = &lB[0][row * LDS_STRIDE + swz(row, kc * 4 + 2 + half) * 8];
    }
  }
  // staging store pointers (buffer 0)
  unsigned short *sAp[4], *sBp[4];
  #pragma unroll
  for (int j = 0; j < 4; ++j) {
    const int row = dq * 4 + j;
    sAp[j] = &lA[0][row * LDS_STRIDE + swz(row, kb) * 8];
    sBp[j] = &lB[0][row * LDS_STRIDE + swz(row, kb) * 8];
  }

  f32x8 acc[2][4];
  #pragma unroll
  for (int i = 0; i < 2; ++i)
    #pragma unroll
    for (int j = 0; j < 4; ++j)
      #pragma unroll
      for (int r = 0; r < 8; ++r) acc[i][j][r] = 0.f;

  f32x4 ra[8], rb[8];
  const float* kp = keys + (size_t)kb * 8 * D_DIM + D0 + dq * 4;
  const float* vp = vals + (size_t)kb * 8 * D_DIM + E0 + dq * 4;

  auto load_stage = [&]() {              // loads next 64-K chunk, advances ptrs
    #pragma unroll
    for (int kk = 0; kk < 8; ++kk) {
      ra[kk] = *(const f32x4*)(kp + (size_t)kk * D_DIM);
      rb[kk] = *(const f32x4*)(vp + (size_t)kk * D_DIM);
    }
    kp += (size_t)64 * D_DIM;
    vp += (size_t)64 * D_DIM;
  };

  auto store_stage = [&](int bufOfs) {   // fp32->bf16 via v_cvt_pk, b128 store
    #pragma unroll
    for (int j = 0; j < 4; ++j) {
      u32x4 pa, pb;
      #pragma unroll
      for (int p = 0; p < 4; ++p) {
        pa[p] = pkbf(ra[2 * p][j], ra[2 * p + 1][j]);
        pb[p] = pkbf(rb[2 * p][j], rb[2 * p + 1][j]);
      }
      *(u32x4*)(sAp[j] + bufOfs) = pa;
      *(u32x4*)(sBp[j] + bufOfs) = pb;
    }
  };

  auto compute = [&](int bufOfs) {
    #pragma unroll
    for (int kc = 0; kc < 2; ++kc) {
      Frag a[2], b[4];
      #pragma unroll
      for (int i = 0; i < 2; ++i) {
        a[i].h[0] = *(const u16x8*)(aP[kc][i][0] + bufOfs);
        a[i].h[1] = *(const u16x8*)(aP[kc][i][1] + bufOfs);
      }
      #pragma unroll
      for (int j = 0; j < 4; ++j) {
        b[j].h[0] = *(const u16x8*)(bP[kc][j][0] + bufOfs);
        b[j].h[1] = *(const u16x8*)(bP[kc][j][1] + bufOfs);
      }
      #pragma unroll
      for (int i = 0; i < 2; ++i)
        #pragma unroll
        for (int j = 0; j < 4; ++j)
          acc[i][j] = __builtin_amdgcn_wmma_f32_16x16x32_bf16(
              false, a[i].bf, false, b[j].bf, (short)0, acc[i][j], false, false);
    }
  };

  const int STAGES = M_DIM / 64;         // 1024 (even)
  load_stage();
  store_stage(0);
  __syncthreads();

  for (int s = 0; s < STAGES - 2; s += 2) {
    load_stage();                        // stage s+1 (global, overlaps WMMA)
    compute(0);                          // stage s   (buf 0)
    __syncthreads();
    store_stage(BUF_OFS);                // stage s+1 -> buf 1
    __syncthreads();

    load_stage();                        // stage s+2
    compute(BUF_OFS);                    // stage s+1 (buf 1)
    __syncthreads();
    store_stage(0);                      // stage s+2 -> buf 0
    __syncthreads();
  }
  load_stage();                          // stage STAGES-1
  compute(0);                            // stage STAGES-2
  __syncthreads();
  store_stage(BUF_OFS);
  __syncthreads();
  compute(BUF_OFS);                      // stage STAGES-1

  // epilogue: C layout VGPR r -> (M = r + half*8, N = L&15); store bf16
  #pragma unroll
  for (int i = 0; i < 2; ++i)
    #pragma unroll
    for (int j = 0; j < 4; ++j)
      #pragma unroll
      for (int r = 0; r < 8; ++r) {
        const int m = D0 + wr * 32 + i * 16 + r + half * 8;
        const int n = E0 + wc * 64 + j * 16 + (L & 15);
        kvb[(size_t)m * D_DIM + n] =
            (unsigned short)(pkbf(acc[i][j][r], acc[i][j][r]) & 0xFFFFu);
      }
}

// ===========================================================================
// Kernel 2: out[s][e] = sum_k x[s][k] * kv[k][e]   (x fp32, kv bf16 from ws)
// grid (64,8), block 256. Same skeleton; 16 K-stages of 64.
// ===========================================================================
__global__ void __launch_bounds__(256)
x_kv_gemm(const float* __restrict__ x,
          const unsigned short* __restrict__ kvb,
          float* __restrict__ out)
{
  __shared__ alignas(16) unsigned short lA[2][BUF_OFS]; // x tile (rows = s)
  __shared__ alignas(16) unsigned short lB[2][BUF_OFS]; // kv^T tile (rows = e)

  const int t    = threadIdx.x;
  const int L    = t & 31;
  const int half = L >> 4;
  const int w    = t >> 5;
  const int wr   = w >> 1;
  const int wc   = w & 1;
  const int S0   = blockIdx.x * 128;
  const int E0   = blockIdx.y * 128;

  // A staging: row sA = t>>1, k-half kh = t&1 (no transpose: k contiguous)
  const int sA = t >> 1;
  const int kh = t & 1;
  // B staging: e-oct eo = t&15, k-quad kq = t>>4 (transpose 4k x 8e)
  const int eo = t & 15;
  const int kq = t >> 4;

  // loop-invariant fragment pointers (buffer 0)
  const unsigned short *aP[2][2][2], *bP[2][4][2];
  #pragma unroll
  for (int kc = 0; kc < 2; ++kc) {
    #pragma unroll
    for (int i = 0; i < 2; ++i) {
      const int row = wr * 32 + i * 16 + (L & 15);
      aP[kc][i][0] = &lA[0][row * LDS_STRIDE + swz(row, kc * 4 + half) * 8];
      aP[kc][i][1] = &lA[0][row * LDS_STRIDE + swz(row, kc * 4 + 2 + half) * 8];
    }
    #pragma unroll
    for (int j = 0; j < 4; ++j) {
      const int row = wc * 64 + j * 16 + (L & 15);
      bP[kc][j][0] = &lB[0][row * LDS_STRIDE + swz(row, kc * 4 + half) * 8];
      bP[kc][j][1] = &lB[0][row * LDS_STRIDE + swz(row, kc * 4 + 2 + half) * 8];
    }
  }
  // staging store pointers (buffer 0)
  unsigned short* sApq[8];
  #pragma unroll
  for (int q = 0; q < 8; ++q) {
    const int k = kh * 32 + q * 4;
    sApq[q] = &lA[0][sA * LDS_STRIDE + swz(sA, k >> 3) * 8 + (k & 7)];
  }
  unsigned short* sBpe[8];
  #pragma unroll
  for (int e = 0; e < 8; ++e) {
    const int row = eo * 8 + e;
    sBpe[e] = &lB[0][row * LDS_STRIDE + swz(row, kq >> 1) * 8 + (kq & 1) * 4];
  }

  f32x8 acc[2][4];
  #pragma unroll
  for (int i = 0; i < 2; ++i)
    #pragma unroll
    for (int j = 0; j < 4; ++j)
      #pragma unroll
      for (int r = 0; r < 8; ++r) acc[i][j][r] = 0.f;

  f32x4 rx[8];
  u16x8 rk[4];
  const float* xp = x + (size_t)(S0 + sA) * D_DIM + kh * 32;
  const unsigned short* bp = kvb + (size_t)(kq * 4) * D_DIM + E0 + eo * 8;

  auto load_stage = [&]() {
    #pragma unroll
    for (int q = 0; q < 8; ++q) rx[q] = *(const f32x4*)(xp + q * 4);
    #pragma unroll
    for (int kk = 0; kk < 4; ++kk) rk[kk] = *(const u16x8*)(bp + (size_t)kk * D_DIM);
    xp += 64;
    bp += (size_t)64 * D_DIM;
  };

  auto store_stage = [&](int bufOfs) {
    #pragma unroll
    for (int q = 0; q < 8; ++q) {              // A: cvt_pk + b64 stores
      u32x2 pk;
      pk[0] = pkbf(rx[q][0], rx[q][1]);
      pk[1] = pkbf(rx[q][2], rx[q][3]);
      *(u32x2*)(sApq[q] + bufOfs) = pk;
    }
    #pragma unroll
    for (int e = 0; e < 8; ++e) {              // B: transpose 4k x 8e
      u16x4 pk;
      #pragma unroll
      for (int kk = 0; kk < 4; ++kk) pk[kk] = rk[kk][e];
      *(u16x4*)(sBpe[e] + bufOfs) = pk;
    }
  };

  auto compute = [&](int bufOfs) {
    #pragma unroll
    for (int kc = 0; kc < 2; ++kc) {
      Frag a[2], b[4];
      #pragma unroll
      for (int i = 0; i < 2; ++i) {
        a[i].h[0] = *(const u16x8*)(aP[kc][i][0] + bufOfs);
        a[i].h[1] = *(const u16x8*)(aP[kc][i][1] + bufOfs);
      }
      #pragma unroll
      for (int j = 0; j < 4; ++j) {
        b[j].h[0] = *(const u16x8*)(bP[kc][j][0] + bufOfs);
        b[j].h[1] = *(const u16x8*)(bP[kc][j][1] + bufOfs);
      }
      #pragma unroll
      for (int i = 0; i < 2; ++i)
        #pragma unroll
        for (int j = 0; j < 4; ++j)
          acc[i][j] = __builtin_amdgcn_wmma_f32_16x16x32_bf16(
              false, a[i].bf, false, b[j].bf, (short)0, acc[i][j], false, false);
    }
  };

  const int STAGES = D_DIM / 64;               // 16 (even)
  load_stage();
  store_stage(0);
  __syncthreads();

  for (int s = 0; s < STAGES - 2; s += 2) {
    load_stage();
    compute(0);
    __syncthreads();
    store_stage(BUF_OFS);
    __syncthreads();

    load_stage();
    compute(BUF_OFS);
    __syncthreads();
    store_stage(0);
    __syncthreads();
  }
  load_stage();
  compute(0);
  __syncthreads();
  store_stage(BUF_OFS);
  __syncthreads();
  compute(BUF_OFS);

  // epilogue: fp32 output
  #pragma unroll
  for (int i = 0; i < 2; ++i)
    #pragma unroll
    for (int j = 0; j < 4; ++j)
      #pragma unroll
      for (int r = 0; r < 8; ++r) {
        const int m = S0 + wr * 32 + i * 16 + r + half * 8;
        const int n = E0 + wc * 64 + j * 16 + (L & 15);
        out[(size_t)m * D_DIM + n] = acc[i][j][r];
      }
}

// ===========================================================================
extern "C" void kernel_launch(void* const* d_in, const int* in_sizes, int n_in,
                              void* d_out, int out_size, void* d_ws, size_t ws_size,
                              hipStream_t stream)
{
  const float* x    = (const float*)d_in[0];
  // d_in[1] is `ver` (==1), unused
  const float* keys = (const float*)d_in[2];
  const float* vals = (const float*)d_in[3];
  float* out = (float*)d_out;
  unsigned short* kvb = (unsigned short*)d_ws;   // 1024*1024 bf16 = 2 MB

  dim3 g1(D_DIM / 128, D_DIM / 128);             // 8 x 8 = 64 blocks
  kv_gemm_bf16<<<g1, 256, 0, stream>>>(keys, vals, kvb);

  dim3 g2(ROWS_X / 128, D_DIM / 128);            // 64 x 8 = 512 blocks
  x_kv_gemm<<<g2, 256, 0, stream>>>(x, kvb, out);
}
